// MambaBlock_13262859010301
// MI455X (gfx1250) — matrix operand order repaired and checked by reading, not compiled, not run on hardware
//
#include <hip/hip_runtime.h>
#include <hip/hip_bf16.h>
#include <math.h>

typedef __bf16 bf16_t;
typedef __attribute__((ext_vector_type(16))) bf16_t v16bf;
typedef __attribute__((ext_vector_type(8)))  bf16_t v8bf;
typedef __attribute__((ext_vector_type(8)))  float  v8f;

#define DMODEL 1024
#define DINNER 4096
#define DSTATE 16
#define DTRANK 64
#define BATCH  2
#define SEQ    2048
#define MROWS  (BATCH*SEQ)   // 4096 token rows
#define XDBLW  96            // dt_rank + 2*d_state

// ---------------------------------------------------------------------------
// WMMA fragment loaders (bf16, 16x16x32), contiguous-k layout used by this code for both A and B.
// A (16x32, MxK): lane (hf,r): elems i -> K = hf*16+i, M = r  (contiguous 32B)
// B (32x16, KxN): lane (hf,r): elems i -> K = hf*16+i, N = r  (contiguous 32B)
// We compute C = A * W^T, so B[k,n] = W[n,k]: both A and W load row-major rows.
// ---------------------------------------------------------------------------
__device__ __forceinline__ v16bf load_fragA(const bf16_t* __restrict__ p, int hf) {
  return *(const v16bf*)(p + hf * 16);
}
__device__ __forceinline__ v16bf load_fragB(const bf16_t* __restrict__ p, int hf) {
  return *(const v16bf*)(p + hf * 16);
}
__device__ __forceinline__ v8f wmma_bf16(v16bf a, v16bf b, v8f c) {
  return __builtin_amdgcn_wmma_f32_16x16x32_bf16(false, a, false, b, (short)0, c,
                                                 false, false);
}

enum { EPI_PLAIN = 0, EPI_CLIP = 1, EPI_RES = 2, EPI_SOFTPLUS = 3 };

// ---------------------------------------------------------------------------
// Generic bf16 WMMA GEMM: C[M,N] = epi( A[M,K] * W[N,K]^T + bias )
// Block = 256 threads = 8 waves; block tile 64(M) x 256(N).
// Each wave: 32x64 sub-tile = 2 A-frags x 4 B-frags -> 8 WMMA per K-step,
// software-pipelined so next K-step's 12 b128 loads are in flight while the
// current 8 v_wmma_f32_16x16x32_bf16 execute.
// ---------------------------------------------------------------------------
template <int EPI>
__global__ __launch_bounds__(256)
void wmma_gemm_bf16(const bf16_t* __restrict__ A, int lda,
                    const bf16_t* __restrict__ W, int ldw,
                    int M, int N, int K,
                    const float* __restrict__ bias,
                    const float* __restrict__ res,
                    float* __restrict__ Cf, bf16_t* __restrict__ Cb, int ldc)
{
  const int lane = threadIdx.x & 31;
  const int wave = threadIdx.x >> 5;
  const int hf   = lane >> 4;      // half-wave select
  const int r    = lane & 15;
  const int wm   = wave >> 2;      // 0..1 -> 32 rows each
  const int wn   = wave & 3;       // 0..3 -> 64 cols each
  const int m0   = blockIdx.y * 64 + wm * 32;
  const int n0   = blockIdx.x * 256 + wn * 64;
  if (m0 >= M || n0 >= N) return;  // wave-uniform exit; EXEC stays all-ones

  int ar0 = m0 + r;      if (ar0 >= M) ar0 = M - 1;
  int ar1 = m0 + 16 + r; if (ar1 >= M) ar1 = M - 1;
  const bf16_t* ap0 = A + (size_t)ar0 * lda;
  const bf16_t* ap1 = A + (size_t)ar1 * lda;
  const bf16_t* wp[4];
  #pragma unroll
  for (int t = 0; t < 4; ++t) {
    int nr = n0 + t * 16 + r; if (nr >= N) nr = N - 1;
    wp[t] = W + (size_t)nr * ldw;
  }

  v8f acc[8] = {};
  // prologue loads (K >= 32 always: smallest K here is 64)
  v16bf a0 = load_fragA(ap0, hf);
  v16bf a1 = load_fragA(ap1, hf);
  v16bf b0 = load_fragB(wp[0], hf);
  v16bf b1 = load_fragB(wp[1], hf);
  v16bf b2 = load_fragB(wp[2], hf);
  v16bf b3 = load_fragB(wp[3], hf);

  for (int k = 0; k < K; k += 32) {
    const int kn = k + 32;
    v16bf a0n = a0, a1n = a1, c0 = b0, c1 = b1, c2 = b2, c3 = b3;
    if (kn < K) {                                 // uniform branch
      if (kn + 128 < K) {                         // near-scope prefetch hint
        __builtin_prefetch((const void*)(ap0 + kn + 128), 0, 3);
        __builtin_prefetch((const void*)(wp[0] + kn + 128), 0, 3);
      }
      a0n = load_fragA(ap0 + kn, hf);
      a1n = load_fragA(ap1 + kn, hf);
      c0  = load_fragB(wp[0] + kn, hf);
      c1  = load_fragB(wp[1] + kn, hf);
      c2  = load_fragB(wp[2] + kn, hf);
      c3  = load_fragB(wp[3] + kn, hf);
    }
    acc[0] = wmma_bf16(a0, b0, acc[0]);
    acc[1] = wmma_bf16(a0, b1, acc[1]);
    acc[2] = wmma_bf16(a0, b2, acc[2]);
    acc[3] = wmma_bf16(a0, b3, acc[3]);
    acc[4] = wmma_bf16(a1, b0, acc[4]);
    acc[5] = wmma_bf16(a1, b1, acc[5]);
    acc[6] = wmma_bf16(a1, b2, acc[6]);
    acc[7] = wmma_bf16(a1, b3, acc[7]);
    a0 = a0n; a1 = a1n; b0 = c0; b1 = c1; b2 = c2; b3 = c3;
  }

  #pragma unroll
  for (int hm = 0; hm < 2; ++hm) {
    #pragma unroll
    for (int t = 0; t < 4; ++t) {
      const int n = n0 + t * 16 + r;
      if (n >= N) continue;
      #pragma unroll
      for (int g = 0; g < 8; ++g) {
        const int m = m0 + hm * 16 + g + 8 * hf;
        if (m >= M) continue;
        float v = acc[hm * 4 + t][g];
        if (bias) v += bias[n];
        if (EPI == EPI_CLIP)     v = fminf(fmaxf(v, -1.f), 1.f);
        if (EPI == EPI_RES)      v += res[(size_t)m * ldc + n];
        if (EPI == EPI_SOFTPLUS) v = (v > 20.f) ? v : log1pf(expf(v));
        if (Cf) Cf[(size_t)m * ldc + n] = v;
        if (Cb) Cb[(size_t)m * ldc + n] = (bf16_t)v;
      }
    }
  }
}

// ---------------------------------------------------------------------------
// LayerNorm over D=1024, one block per row, output bf16 (GEMM A operand).
// ---------------------------------------------------------------------------
__global__ __launch_bounds__(256)
void layernorm_to_bf16(const float* __restrict__ x, const float* __restrict__ g,
                       const float* __restrict__ b, bf16_t* __restrict__ out, int D)
{
  const int row = blockIdx.x;
  const float* xr = x + (size_t)row * D;
  __shared__ float s1[256], s2[256];
  float sum = 0.f, sq = 0.f;
  for (int i = threadIdx.x; i < D; i += 256) { float v = xr[i]; sum += v; sq += v * v; }
  s1[threadIdx.x] = sum; s2[threadIdx.x] = sq;
  __syncthreads();
  for (int off = 128; off > 0; off >>= 1) {
    if (threadIdx.x < off) { s1[threadIdx.x] += s1[threadIdx.x + off];
                             s2[threadIdx.x] += s2[threadIdx.x + off]; }
    __syncthreads();
  }
  const float mu  = s1[0] / (float)D;
  const float var = s2[0] / (float)D - mu * mu;
  const float rs  = rsqrtf(var + 1e-5f);
  bf16_t* outr = out + (size_t)row * D;
  for (int i = threadIdx.x; i < D; i += 256)
    outr[i] = (bf16_t)((xr[i] - mu) * rs * g[i] + b[i]);
}

__global__ void f32_to_bf16(const float* __restrict__ in, bf16_t* __restrict__ out, size_t n)
{
  size_t i = (size_t)blockIdx.x * blockDim.x + threadIdx.x;
  size_t stride = (size_t)gridDim.x * blockDim.x;
  for (; i < n; i += stride) out[i] = (bf16_t)in[i];
}

// causal depthwise conv (D_CONV=4) + SiLU on u = xz[:, :DINNER]
__global__ __launch_bounds__(256)
void conv_silu(const float* __restrict__ xz, const float* __restrict__ cw,
               const float* __restrict__ cb, bf16_t* __restrict__ ub,
               float* __restrict__ uf)
{
  size_t idx = (size_t)blockIdx.x * 256 + threadIdx.x;
  if (idx >= (size_t)MROWS * DINNER) return;
  const int d = (int)(idx & (DINNER - 1));
  const int m = (int)(idx >> 12);      // token row = b*SEQ + l
  const int l = m & (SEQ - 1);
  float acc = cb[d];
  #pragma unroll
  for (int j = 0; j < 4; ++j) {
    int ll = l - 3 + j;
    if (ll >= 0)
      acc += cw[d * 4 + j] * xz[(size_t)(m - 3 + j) * (2 * DINNER) + d];
  }
  float s = acc / (1.f + expf(-acc));  // silu
  ub[idx] = (bf16_t)s;
  uf[idx] = s;
}

__global__ void extract_dt(const float* __restrict__ xdbl, bf16_t* __restrict__ dtb)
{
  int i = blockIdx.x * 256 + threadIdx.x;
  if (i >= MROWS * DTRANK) return;
  int m = i >> 6, c = i & 63;
  dtb[i] = (bf16_t)xdbl[(size_t)m * XDBLW + c];
}

// selective scan: thread = one (b, d) channel; 16-entry state in registers;
// B_t / C_t broadcast through double-buffered LDS (one barrier per step).
__global__ __launch_bounds__(256)
void selective_scan(const float* __restrict__ dtf, const float* __restrict__ xdbl,
                    const float* __restrict__ uf, const float* __restrict__ A_log,
                    float* __restrict__ ys)
{
  const int b = blockIdx.y;
  const int d = blockIdx.x * 256 + threadIdx.x;
  __shared__ float sBC[2][2 * DSTATE];
  float Ad[DSTATE], h[DSTATE];
  #pragma unroll
  for (int s = 0; s < DSTATE; ++s) { Ad[s] = -__expf(A_log[(size_t)d * DSTATE + s]); h[s] = 0.f; }
  // preload l = 0
  if (threadIdx.x < 2 * DSTATE)
    sBC[0][threadIdx.x] = xdbl[((size_t)b * SEQ) * XDBLW + DTRANK + threadIdx.x];
  for (int l = 0; l < SEQ; ++l) {
    const size_t row = (size_t)b * SEQ + l;
    const int cur = l & 1;
    __syncthreads();   // sBC[cur] ready; sBC[cur^1] free to overwrite
    if (threadIdx.x < 2 * DSTATE && l + 1 < SEQ)
      sBC[cur ^ 1][threadIdx.x] = xdbl[(row + 1) * XDBLW + DTRANK + threadIdx.x];
    const float dt = dtf[row * DINNER + d];
    const float du = dt * uf[row * DINNER + d];
    float y = 0.f;
    #pragma unroll
    for (int s = 0; s < DSTATE; ++s) {
      float hv = h[s] * __expf(dt * Ad[s]) + du * sBC[cur][s];
      h[s] = hv;
      y += hv * sBC[cur][DSTATE + s];
    }
    ys[row * DINNER + d] = y;
  }
}

// y = (ys + u*D_skip) * silu(z) -> bf16 (out_proj A operand)
__global__ __launch_bounds__(256)
void gate_mul(const float* __restrict__ ys, const float* __restrict__ uf,
              const float* __restrict__ Dsk, const float* __restrict__ xz,
              bf16_t* __restrict__ yb)
{
  size_t idx = (size_t)blockIdx.x * 256 + threadIdx.x;
  if (idx >= (size_t)MROWS * DINNER) return;
  const int d = (int)(idx & (DINNER - 1));
  const size_t m = idx >> 12;
  const float z = xz[m * (2 * DINNER) + DINNER + d];
  const float sz = z / (1.f + expf(-z));
  yb[idx] = (bf16_t)((ys[idx] + uf[idx] * Dsk[d]) * sz);
}

// ---------------------------------------------------------------------------
extern "C" void kernel_launch(void* const* d_in, const int* in_sizes, int n_in,
                              void* d_out, int out_size, void* d_ws, size_t ws_size,
                              hipStream_t stream)
{
  (void)in_sizes; (void)n_in; (void)out_size; (void)ws_size;
  const float* x        = (const float*)d_in[0];
  const float* ln1_g    = (const float*)d_in[1];
  const float* ln1_b    = (const float*)d_in[2];
  const float* w1       = (const float*)d_in[3];
  const float* b1       = (const float*)d_in[4];
  const float* w2       = (const float*)d_in[5];
  const float* b2       = (const float*)d_in[6];
  const float* ln2_g    = (const float*)d_in[7];
  const float* ln2_b    = (const float*)d_in[8];
  const float* in_proj  = (const float*)d_in[9];
  const float* conv_w   = (const float*)d_in[10];
  const float* conv_b   = (const float*)d_in[11];
  const float* x_proj   = (const float*)d_in[12];
  const float* dt_proj  = (const float*)d_in[13];
  const float* dt_projb = (const float*)d_in[14];
  const float* A_log    = (const float*)d_in[15];
  const float* D_skip   = (const float*)d_in[16];
  const float* out_proj = (const float*)d_in[17];

  char* base = (char*)d_ws;
  size_t off = 0;
  auto alloc = [&](size_t bytes) -> void* {
    void* r = base + off;
    off += (bytes + 255) & ~(size_t)255;
    return r;
  };
  bf16_t* w1b = (bf16_t*)alloc((size_t)DINNER * DMODEL * 2);
  bf16_t* w2b = (bf16_t*)alloc((size_t)DMODEL * DINNER * 2);
  bf16_t* ipb = (bf16_t*)alloc((size_t)2 * DINNER * DMODEL * 2);
  bf16_t* xpb = (bf16_t*)alloc((size_t)XDBLW * DINNER * 2);
  bf16_t* dpb = (bf16_t*)alloc((size_t)DINNER * DTRANK * 2);
  bf16_t* opb = (bf16_t*)alloc((size_t)DMODEL * DINNER * 2);
  bf16_t* lnb = (bf16_t*)alloc((size_t)MROWS * DMODEL * 2);   // LN1 out, reused for LN2
  bf16_t* gtb = (bf16_t*)alloc((size_t)MROWS * DINNER * 2);   // gated bf16, reused for y bf16
  float*  x2  = (float*) alloc((size_t)MROWS * DMODEL * 4);
  float*  xz  = (float*) alloc((size_t)MROWS * 2 * DINNER * 4);
  bf16_t* ucb = (bf16_t*)alloc((size_t)MROWS * DINNER * 2);
  float*  ucf = (float*) alloc((size_t)MROWS * DINNER * 4);
  float*  xdb = (float*) alloc((size_t)MROWS * XDBLW * 4);
  bf16_t* dtb = (bf16_t*)alloc((size_t)MROWS * DTRANK * 2);
  float*  dtf = (float*) alloc((size_t)MROWS * DINNER * 4);
  float*  ys  = (float*) alloc((size_t)MROWS * DINNER * 4);

  auto cvt = [&](const float* src, bf16_t* dst, size_t n) {
    f32_to_bf16<<<(int)((n + 255) / 256), 256, 0, stream>>>(src, dst, n);
  };
  cvt(w1, w1b, (size_t)DINNER * DMODEL);
  cvt(w2, w2b, (size_t)DMODEL * DINNER);
  cvt(in_proj, ipb, (size_t)2 * DINNER * DMODEL);
  cvt(x_proj, xpb, (size_t)XDBLW * DINNER);
  cvt(dt_proj, dpb, (size_t)DINNER * DTRANK);
  cvt(out_proj, opb, (size_t)DMODEL * DINNER);

  auto gemm_grid = [](int Mv, int Nv) { return dim3((Nv + 255) / 256, (Mv + 63) / 64); };
  const dim3 blk(256);

  // FFN: gated = clip(LN1(x) @ w1^T + b1), x2 = x + gated @ w2^T + b2
  layernorm_to_bf16<<<MROWS, 256, 0, stream>>>(x, ln1_g, ln1_b, lnb, DMODEL);
  wmma_gemm_bf16<EPI_CLIP><<<gemm_grid(MROWS, DINNER), blk, 0, stream>>>(
      lnb, DMODEL, w1b, DMODEL, MROWS, DINNER, DMODEL, b1, nullptr, nullptr, gtb, DINNER);
  wmma_gemm_bf16<EPI_RES><<<gemm_grid(MROWS, DMODEL), blk, 0, stream>>>(
      gtb, DINNER, w2b, DINNER, MROWS, DMODEL, DINNER, b2, x, x2, nullptr, DMODEL);

  // Mamba: xz = LN2(x2) @ in_proj^T
  layernorm_to_bf16<<<MROWS, 256, 0, stream>>>(x2, ln2_g, ln2_b, lnb, DMODEL);
  wmma_gemm_bf16<EPI_PLAIN><<<gemm_grid(MROWS, 2 * DINNER), blk, 0, stream>>>(
      lnb, DMODEL, ipb, DMODEL, MROWS, 2 * DINNER, DMODEL, nullptr, nullptr, xz, nullptr, 2 * DINNER);

  {
    size_t n = (size_t)MROWS * DINNER;
    conv_silu<<<(int)((n + 255) / 256), 256, 0, stream>>>(xz, conv_w, conv_b, ucb, ucf);
  }

  // x_dbl = u @ x_proj^T (N=96, bounds-clamped tiles)
  wmma_gemm_bf16<EPI_PLAIN><<<gemm_grid(MROWS, XDBLW), blk, 0, stream>>>(
      ucb, DINNER, xpb, DINNER, MROWS, XDBLW, DINNER, nullptr, nullptr, xdb, nullptr, XDBLW);

  extract_dt<<<(MROWS * DTRANK + 255) / 256, 256, 0, stream>>>(xdb, dtb);

  // dt_full = softplus(dt @ dt_proj^T + dt_proj_b)
  wmma_gemm_bf16<EPI_SOFTPLUS><<<gemm_grid(MROWS, DINNER), blk, 0, stream>>>(
      dtb, DTRANK, dpb, DTRANK, MROWS, DINNER, DTRANK, dt_projb, nullptr, dtf, nullptr, DINNER);

  selective_scan<<<dim3(DINNER / 256, BATCH), 256, 0, stream>>>(dtf, xdb, ucf, A_log, ys);

  {
    size_t n = (size_t)MROWS * DINNER;
    gate_mul<<<(int)((n + 255) / 256), 256, 0, stream>>>(ys, ucf, D_skip, xz, gtb);
  }

  // out = x2 + y @ out_proj^T
  wmma_gemm_bf16<EPI_RES><<<gemm_grid(MROWS, DMODEL), blk, 0, stream>>>(
      gtb, DINNER, opb, DINNER, MROWS, DMODEL, DINNER, nullptr, x2, (float*)d_out, nullptr, DMODEL);
}
